// LN_miLSTM_37374805410200
// MI455X (gfx1250) — compile-verified, matrix operand order checked by reading, not run on hardware
//
#include <hip/hip_runtime.h>
#include <hip/hip_bf16.h>
#include <math.h>

#define B_SZ 1024
#define H_SZ 2048
#define E_SZ 2048
#define O_SZ 50257
#define EPS  1e-5f

typedef __attribute__((ext_vector_type(16))) __bf16 v16bf;
typedef __attribute__((ext_vector_type(8)))  __bf16 v8bf;
typedef __attribute__((ext_vector_type(8)))  float  v8f;

// ---------------------------------------------------------------------------
// Tiled bf16 WMMA GEMM:  C[M,N] = A[M,K] @ W[N,K]^T + bias
//   - gather != nullptr: row m of A = Aemb[gather[m]*K ...] (embedding fuse).
//   - M multiple of 128, K multiple of 64 (true here: 1024 / 2048). N ragged ok.
// Block: 256 threads = 8 waves. Macro-tile 128(M) x 64(N).
// Wave grid 4(M) x 2(N); each wave owns a 2x2 grid of 16x16 tiles
// -> 4 v_wmma_f32_16x16x32_bf16 per K-step, each A/B LDS fragment reused 2x.
// Depth-2 software pipeline, unrolled x2 so every pipeline index is a
// compile-time constant (two named register sets, LDS buffer by pointer):
// no dynamic local-array indexing -> no scratch spills. Each tile's loadcnt
// wait lands a full iteration (WMMA phase + barrier) after its loads issue.
// ---------------------------------------------------------------------------
__global__ __launch_bounds__(256)
void gemm_bf16_wmma(const float* __restrict__ A,
                    const int*   __restrict__ gather,
                    const float* __restrict__ Aemb,
                    const float* __restrict__ W,
                    const float* __restrict__ bias,
                    float* __restrict__ C,
                    int M, int N, int K)
{
    // Double-buffered LDS tiles, bf16, padded row stride (40 elems = 80B).
    __shared__ __align__(16) __bf16 As[2][128][40];
    __shared__ __align__(16) __bf16 Bs[2][64][40];

    const int t     = threadIdx.x;
    const int wave  = t >> 5;
    const int lane  = t & 31;
    const int waveM = wave & 3;          // 0..3 -> 32-row band
    const int waveN = wave >> 2;         // 0..1 -> 32-col band
    const int lrow  = lane & 15;
    const int lhi   = lane >> 4;         // 0: K{0..7,16..23}, 1: K{8..15,24..31}

    const int tileM = blockIdx.x * 128;  // x walks M so same-N blocks are
    const int tileN = blockIdx.y * 64;   // adjacent -> Wdec slice stays in L2

    // ---- hoisted staging pointers (gather resolved ONCE) -------------------
    const int r0 = t >> 3;               // 0..31
    const int c4 = (t & 7) * 4;          // 0,4,..,28
    const float* aptr[4];
    #pragma unroll
    for (int i = 0; i < 4; ++i) {
        const int gm = tileM + r0 + 32 * i;
        aptr[i] = gather ? (Aemb + (size_t)gather[gm] * K + c4)
                         : (A    + (size_t)gm        * K + c4);
    }
    const float* bptr[2];
    bool bvalid[2];
    #pragma unroll
    for (int i = 0; i < 2; ++i) {
        const int gn = tileN + r0 + 32 * i;
        bvalid[i] = (gn < N);
        bptr[i]   = W + (size_t)(bvalid[i] ? gn : 0) * K + c4;  // clamped: safe
    }

    // Two statically-named register sets (constant indices only -> no spill).
    float4 a0[4], b0[2], a1[4], b1[2];

    auto load_set = [&](float4 (&ar)[4], float4 (&br)[2], int k0) {
        #pragma unroll
        for (int i = 0; i < 4; ++i)
            ar[i] = *(const float4*)(aptr[i] + k0);
        #pragma unroll
        for (int i = 0; i < 2; ++i) {
            const float4 v = *(const float4*)(bptr[i] + k0);  // unconditional
            br[i].x = bvalid[i] ? v.x : 0.f;                  // select, no exec
            br[i].y = bvalid[i] ? v.y : 0.f;
            br[i].z = bvalid[i] ? v.z : 0.f;
            br[i].w = bvalid[i] ? v.w : 0.f;
        }
    };
    auto store_set = [&](const float4 (&ar)[4], const float4 (&br)[2],
                         __bf16 (*Asb)[40], __bf16 (*Bsb)[40]) {
        #pragma unroll
        for (int i = 0; i < 4; ++i) {
            __bf16* d = &Asb[r0 + 32 * i][c4];
            d[0] = (__bf16)ar[i].x; d[1] = (__bf16)ar[i].y;
            d[2] = (__bf16)ar[i].z; d[3] = (__bf16)ar[i].w;
        }
        #pragma unroll
        for (int i = 0; i < 2; ++i) {
            __bf16* d = &Bsb[r0 + 32 * i][c4];
            d[0] = (__bf16)br[i].x; d[1] = (__bf16)br[i].y;
            d[2] = (__bf16)br[i].z; d[3] = (__bf16)br[i].w;
        }
    };

    v8f acc[2][2];
    #pragma unroll
    for (int mi = 0; mi < 2; ++mi)
        #pragma unroll
        for (int ni = 0; ni < 2; ++ni)
            acc[mi][ni] = (v8f){0.f, 0.f, 0.f, 0.f, 0.f, 0.f, 0.f, 0.f};

    // 4 WMMAs from one LDS buffer (ISA 16-bit layout: lanes 0-15 K{0..7,16..23}).
    auto compute = [&](__bf16 (*Asb)[40], __bf16 (*Bsb)[40]) {
        v16bf af[2], bf[2];
        #pragma unroll
        for (int mi = 0; mi < 2; ++mi) {
            const __bf16* p = &Asb[waveM * 32 + mi * 16 + lrow][lhi * 8];
            const v8bf lo = *(const v8bf*)(p);
            const v8bf hi = *(const v8bf*)(p + 16);
            #pragma unroll
            for (int q = 0; q < 8; ++q) { af[mi][q] = lo[q]; af[mi][q + 8] = hi[q]; }
        }
        #pragma unroll
        for (int ni = 0; ni < 2; ++ni) {
            const __bf16* p = &Bsb[waveN * 32 + ni * 16 + lrow][lhi * 8];
            const v8bf lo = *(const v8bf*)(p);
            const v8bf hi = *(const v8bf*)(p + 16);
            #pragma unroll
            for (int q = 0; q < 8; ++q) { bf[ni][q] = lo[q]; bf[ni][q + 8] = hi[q]; }
        }
        #pragma unroll
        for (int mi = 0; mi < 2; ++mi)
            #pragma unroll
            for (int ni = 0; ni < 2; ++ni)
                acc[mi][ni] = __builtin_amdgcn_wmma_f32_16x16x32_bf16(
                    false, af[mi], false, bf[ni], (short)0, acc[mi][ni],
                    false, false);
    };

    // ---- prologue: tiles 0 and 1 into registers, tile 0 into LDS[0] --------
    load_set(a0, b0, 0);
    load_set(a1, b1, 32);
    store_set(a0, b0, As[0], Bs[0]);     // in-order loads: waits set 0 only

    // ---- main loop: two K-steps per trip, all pipeline indices constant ----
    for (int k0 = 0; k0 < K; k0 += 64) {
        const bool more = (k0 + 64 < K);

        if (more) load_set(a0, b0, k0 + 64);     // tile +2 in flight
        __syncthreads();
        compute(As[0], Bs[0]);                   // tile k0/32
        store_set(a1, b1, As[1], Bs[1]);         // tile +1 -> LDS[1]

        if (more) load_set(a1, b1, k0 + 96);     // tile +3 in flight
        __syncthreads();
        compute(As[1], Bs[1]);                   // tile k0/32 + 1
        if (more) store_set(a0, b0, As[0], Bs[0]); // tile +2 -> LDS[0]
    }

    // ---- epilogue: lane holds column n; rows m = lhi*8 + r -----------------
    #pragma unroll
    for (int ni = 0; ni < 2; ++ni) {
        const int gn = tileN + waveN * 32 + ni * 16 + lrow;
        if (gn < N) {
            const float bv = bias ? bias[gn] : 0.f;
            #pragma unroll
            for (int mi = 0; mi < 2; ++mi) {
                const int gm0 = tileM + waveM * 32 + mi * 16 + lhi * 8;
                #pragma unroll
                for (int r = 0; r < 8; ++r)
                    C[(size_t)(gm0 + r) * N + gn] = acc[mi][ni][r] + bv;
            }
        }
    }
}

// ---------------------------------------------------------------------------
// Row LayerNorm (torch-style: unbiased var, /(std+eps)) applied in place to
// the 8 gate pre-activation buffers. grid = (B, 8); g<4 -> lnx, g>=4 -> lnh.
// ---------------------------------------------------------------------------
__global__ __launch_bounds__(256)
void ln_rows(float* __restrict__ gates,
             const float* __restrict__ lnx_a, const float* __restrict__ lnx_b,
             const float* __restrict__ lnh_a, const float* __restrict__ lnh_b)
{
    const int b = blockIdx.x;
    const int g = blockIdx.y;
    float* __restrict__ row = gates + ((size_t)g * B_SZ + b) * H_SZ;
    const float* __restrict__ ga = (g < 4) ? lnx_a : lnh_a;
    const float* __restrict__ gb = (g < 4) ? lnx_b : lnh_b;

    __shared__ float rs[8], rs2[8];
    float s = 0.f, s2 = 0.f;
    #pragma unroll
    for (int j = 0; j < H_SZ / 256; ++j) {
        const float z = row[threadIdx.x + j * 256];
        s += z; s2 += z * z;
    }
    #pragma unroll
    for (int o = 16; o > 0; o >>= 1) {
        s  += __shfl_xor(s,  o, 32);
        s2 += __shfl_xor(s2, o, 32);
    }
    if ((threadIdx.x & 31) == 0) { rs[threadIdx.x >> 5] = s; rs2[threadIdx.x >> 5] = s2; }
    __syncthreads();
    float ts = 0.f, ts2 = 0.f;
    #pragma unroll
    for (int i = 0; i < 8; ++i) { ts += rs[i]; ts2 += rs2[i]; }

    const float mu  = ts / (float)H_SZ;
    const float var = (ts2 - (float)H_SZ * mu * mu) / (float)(H_SZ - 1);
    const float inv = 1.f / (sqrtf(var) + EPS);

    #pragma unroll
    for (int j = 0; j < H_SZ / 256; ++j) {
        const int h = threadIdx.x + j * 256;
        row[h] = (row[h] - mu) * inv * ga[h] + gb[h];
    }
}

// ---------------------------------------------------------------------------
// Gate fusion + LN(cx) + hx. One block per batch row; writes cx/hx directly
// into the tail of d_out.
// ---------------------------------------------------------------------------
__global__ __launch_bounds__(256)
void mi_fuse(const float* __restrict__ gates, const float* __restrict__ c0,
             const float* __restrict__ af, const float* __restrict__ bf1, const float* __restrict__ bf2,
             const float* __restrict__ ai, const float* __restrict__ bi1, const float* __restrict__ bi2,
             const float* __restrict__ ao, const float* __restrict__ bo1, const float* __restrict__ bo2,
             const float* __restrict__ az, const float* __restrict__ bz1, const float* __restrict__ bz2,
             const float* __restrict__ lno_a, const float* __restrict__ lno_b,
             float* __restrict__ hx, float* __restrict__ cx)
{
    const int b = blockIdx.x;
    const size_t BH = (size_t)B_SZ * H_SZ;
    const size_t ro = (size_t)b * H_SZ;
    const float* Fx = gates + 0 * BH + ro;
    const float* Ix = gates + 1 * BH + ro;
    const float* Zx = gates + 2 * BH + ro;
    const float* Ox = gates + 3 * BH + ro;
    const float* Fh = gates + 4 * BH + ro;
    const float* Ih = gates + 5 * BH + ro;
    const float* Zh = gates + 6 * BH + ro;
    const float* Oh = gates + 7 * BH + ro;

    float cv[H_SZ / 256], ov[H_SZ / 256];
    float s = 0.f, s2 = 0.f;
    #pragma unroll
    for (int j = 0; j < H_SZ / 256; ++j) {
        const int h = threadIdx.x + j * 256;
        const float fx = Fx[h], ih_ = Ih[h], zx = Zx[h], ox = Ox[h];
        const float fh = Fh[h], ix = Ix[h], zh = Zh[h], oh = Oh[h];
        const float fg = 1.f / (1.f + __expf(-(af[h] * fx * fh + bf1[h] * fx + bf2[h] * fh)));
        const float ig = 1.f / (1.f + __expf(-(ai[h] * ix * ih_ + bi1[h] * ix + bi2[h] * ih_)));
        const float og = 1.f / (1.f + __expf(-(ao[h] * ox * oh + bo1[h] * ox + bo2[h] * oh)));
        const float zt = tanhf(az[h] * zx * zh + bz1[h] * zx + bz2[h] * zh);
        const float c  = fg * c0[ro + h] + ig * zt;
        cv[j] = c; ov[j] = og;
        cx[ro + h] = c;
        s += c; s2 += c * c;
    }
    __shared__ float rs[8], rs2[8];
    #pragma unroll
    for (int o = 16; o > 0; o >>= 1) {
        s  += __shfl_xor(s,  o, 32);
        s2 += __shfl_xor(s2, o, 32);
    }
    if ((threadIdx.x & 31) == 0) { rs[threadIdx.x >> 5] = s; rs2[threadIdx.x >> 5] = s2; }
    __syncthreads();
    float ts = 0.f, ts2 = 0.f;
    #pragma unroll
    for (int i = 0; i < 8; ++i) { ts += rs[i]; ts2 += rs2[i]; }
    const float mu  = ts / (float)H_SZ;
    const float var = (ts2 - (float)H_SZ * mu * mu) / (float)(H_SZ - 1);
    const float inv = 1.f / (sqrtf(var) + EPS);

    #pragma unroll
    for (int j = 0; j < H_SZ / 256; ++j) {
        const int h = threadIdx.x + j * 256;
        const float n = (cv[j] - mu) * inv * lno_a[h] + lno_b[h];
        hx[ro + h] = ov[j] * tanhf(n);
    }
}

// ---------------------------------------------------------------------------
extern "C" void kernel_launch(void* const* d_in, const int* in_sizes, int n_in,
                              void* d_out, int out_size, void* d_ws, size_t ws_size,
                              hipStream_t stream)
{
    const int*   inp  = (const int*)  d_in[0];
    const float* h0   = (const float*)d_in[1];
    const float* c0   = (const float*)d_in[2];
    const float* emb  = (const float*)d_in[3];
    const float* Wfx  = (const float*)d_in[4];  const float* bfx = (const float*)d_in[5];
    const float* Wix  = (const float*)d_in[6];  const float* bix = (const float*)d_in[7];
    const float* Wzx  = (const float*)d_in[8];  const float* bzx = (const float*)d_in[9];
    const float* Wox  = (const float*)d_in[10]; const float* box_= (const float*)d_in[11];
    const float* Wfh  = (const float*)d_in[12]; const float* bfh = (const float*)d_in[13];
    const float* Wih  = (const float*)d_in[14]; const float* bih = (const float*)d_in[15];
    const float* Wzh  = (const float*)d_in[16]; const float* bzh = (const float*)d_in[17];
    const float* Woh  = (const float*)d_in[18]; const float* boh = (const float*)d_in[19];
    const float* af   = (const float*)d_in[20]; const float* bf1 = (const float*)d_in[21]; const float* bf2 = (const float*)d_in[22];
    const float* ai   = (const float*)d_in[23]; const float* bi1 = (const float*)d_in[24]; const float* bi2 = (const float*)d_in[25];
    const float* ao   = (const float*)d_in[26]; const float* bo1 = (const float*)d_in[27]; const float* bo2 = (const float*)d_in[28];
    const float* az   = (const float*)d_in[29]; const float* bz1 = (const float*)d_in[30]; const float* bz2 = (const float*)d_in[31];
    const float* lnx_a = (const float*)d_in[32]; const float* lnx_b = (const float*)d_in[33];
    const float* lnh_a = (const float*)d_in[34]; const float* lnh_b = (const float*)d_in[35];
    const float* lno_a = (const float*)d_in[36]; const float* lno_b = (const float*)d_in[37];
    const float* Wdec  = (const float*)d_in[38]; const float* bdec  = (const float*)d_in[39];

    float* out   = (float*)d_out;                      // [B, O]
    const size_t BH = (size_t)B_SZ * H_SZ;
    float* hx    = out + (size_t)B_SZ * O_SZ;          // [B, H] (output #2)
    float* cx    = hx + BH;                            // [B, H] (output #3)
    float* gates = (float*)d_ws;                       // 8 * B * H floats (64 MB)

    const dim3 blk(256);
    const dim3 ggate(B_SZ / 128, H_SZ / 64);           // (M tiles, N tiles)

    // x-side gate pre-activations (embedding gather fused into A staging)
    gemm_bf16_wmma<<<ggate, blk, 0, stream>>>(nullptr, inp, emb, Wfx, bfx, gates + 0 * BH, B_SZ, H_SZ, E_SZ);
    gemm_bf16_wmma<<<ggate, blk, 0, stream>>>(nullptr, inp, emb, Wix, bix, gates + 1 * BH, B_SZ, H_SZ, E_SZ);
    gemm_bf16_wmma<<<ggate, blk, 0, stream>>>(nullptr, inp, emb, Wzx, bzx, gates + 2 * BH, B_SZ, H_SZ, E_SZ);
    gemm_bf16_wmma<<<ggate, blk, 0, stream>>>(nullptr, inp, emb, Wox, box_, gates + 3 * BH, B_SZ, H_SZ, E_SZ);
    // h-side gate pre-activations
    gemm_bf16_wmma<<<ggate, blk, 0, stream>>>(h0, nullptr, nullptr, Wfh, bfh, gates + 4 * BH, B_SZ, H_SZ, H_SZ);
    gemm_bf16_wmma<<<ggate, blk, 0, stream>>>(h0, nullptr, nullptr, Wih, bih, gates + 5 * BH, B_SZ, H_SZ, H_SZ);
    gemm_bf16_wmma<<<ggate, blk, 0, stream>>>(h0, nullptr, nullptr, Wzh, bzh, gates + 6 * BH, B_SZ, H_SZ, H_SZ);
    gemm_bf16_wmma<<<ggate, blk, 0, stream>>>(h0, nullptr, nullptr, Woh, boh, gates + 7 * BH, B_SZ, H_SZ, H_SZ);

    // LayerNorm all 8 gate buffers in place
    ln_rows<<<dim3(B_SZ, 8), blk, 0, stream>>>(gates, lnx_a, lnx_b, lnh_a, lnh_b);

    // Multiplicative-integration gating + LN(cx) -> hx, cx
    mi_fuse<<<dim3(B_SZ), blk, 0, stream>>>(gates, c0,
                                            af, bf1, bf2, ai, bi1, bi2,
                                            ao, bo1, bo2, az, bz1, bz2,
                                            lno_a, lno_b, hx, cx);

    // Decoder: out = hx @ Wdec^T + bdec   (N = 50257, ragged tiles guarded;
    // blockIdx.x walks M so the Wdec slice is reused out of L2)
    gemm_bf16_wmma<<<dim3(B_SZ / 128, (O_SZ + 63) / 64), blk, 0, stream>>>(
        hx, nullptr, nullptr, Wdec, bdec, out, B_SZ, O_SZ, H_SZ);
}
